// DCTFrequencyEncoder_26276609917498
// MI455X (gfx1250) — matrix operand order repaired and checked by reading, not compiled, not run on hardware
//
#include <hip/hip_runtime.h>
#include <hip/hip_bf16.h>

// ---------------- CDNA5 types ----------------
typedef __attribute__((ext_vector_type(16))) __bf16 v16bf;
typedef __attribute__((ext_vector_type(8)))  float  v8f;
typedef __attribute__((ext_vector_type(4)))  int    i4;   // trivial 16B vector (union-safe)

#define HH    224
#define WW    224
#define KHM   112      // masked DCT rows kept
#define KWM   112
#define BATCH 32
#define C1    64
#define C2    128
#define EMB   256
// conv2 WMMA region A: y in [0,115), x in [0,128); rest handled analytically.
#define YROWS  115
#define XTILES 8
#define RPW    4       // output rows per wave (4 accumulators share one B fragment)
// h1 is only needed on [0,116) x [0,129) (A-region + 1-px halo)
#define H1H   116
#define H1W   129

__device__ __forceinline__ unsigned short f2bf(float f) {
  unsigned int u = __float_as_uint(f);
  u += 0x7FFFu + ((u >> 16) & 1u);      // round-to-nearest-even
  return (unsigned short)(u >> 16);
}

// ---- CDNA5 async global->LDS DMA (ASYNCcnt-tracked), CDNA5 ISA ----
__device__ __forceinline__ void async_b128(unsigned int lds_off, const void* gptr) {
  asm volatile("global_load_async_to_lds_b128 %0, %1, off"
               :: "v"(lds_off), "v"(gptr) : "memory");
}
__device__ __forceinline__ void wait_async0() {
  asm volatile("s_wait_asynccnt 0" ::: "memory");
}

// ---------------- DCT basis: Bk[k][n] = 2*cos(pi*k*(2n+1)/(2N)), k<112 only (mask!) ----
__global__ void basis_kernel(float* __restrict__ Bk) {
  int id = blockIdx.x * blockDim.x + threadIdx.x;
  if (id >= KHM * WW) return;
  int k = id / WW, n = id % WW;
  int t = (k * (2 * n + 1)) % (4 * WW);          // reduce mod period -> accurate cosf
  float arg = 3.14159265358979323846f * (float)t / (float)(2 * WW);
  Bk[id] = 2.0f * cosf(arg);
}

// ---------------- DCT stage A: Y[b,h,k] = sum_w x[b,h,w]*Bk[k,w]  (k<112) ----------
__global__ void dctA_kernel(const float* __restrict__ x, const float* __restrict__ Bk,
                            float* __restrict__ Y) {
  int id = blockIdx.x * blockDim.x + threadIdx.x;
  if (id >= BATCH * HH * KWM) return;
  int k = id % KWM;
  int bh = id / KWM;                              // b*224 + h
  const float* xr = x + (size_t)bh * WW;
  const float* br = Bk + (size_t)k * WW;
  float s = 0.f;
  for (int w = 0; w < WW; ++w) s = fmaf(xr[w], br[w], s);
  Y[(size_t)bh * KWM + k] = s;
}

// ---- DCT stage B + mask: X[b,kh,kw], only the [0,117)x[0,130) region conv1 reads ----
__global__ void dctB_kernel(const float* __restrict__ Y, const float* __restrict__ Bk,
                            float* __restrict__ X) {
  int id = blockIdx.x * blockDim.x + threadIdx.x;
  if (id >= BATCH * (H1H + 1) * (H1W + 1)) return;
  int kw = id % (H1W + 1);
  int r  = id / (H1W + 1);
  int kh = r % (H1H + 1);
  int b  = r / (H1H + 1);
  float s = 0.f;
  if (kh < KHM && kw < KWM) {
    const float* br = Bk + (size_t)kh * WW;       // over h
    const float* yc = Y + (size_t)b * HH * KWM + kw;
    for (int h = 0; h < HH; ++h) s = fmaf(br[h], yc[(size_t)h * KWM], s);
  }
  X[((size_t)(b * HH + kh)) * WW + kw] = s;       // full-stride buffer, subset written
}

// ---------------- fold BN into scale/shift: sc = [s1(64) t1(64) s2(128) t2(128)] ----------
__global__ void scales_kernel(const float* b1, const float* g1, const float* be1,
                              const float* m1, const float* v1,
                              const float* b2, const float* g2, const float* be2,
                              const float* m2, const float* v2, float* __restrict__ sc) {
  int c = threadIdx.x;
  if (c < C1) {
    float s = g1[c] * rsqrtf(v1[c] + 1e-5f);
    sc[c] = s;
    sc[C1 + c] = (b1[c] - m1[c]) * s + be1[c];
  } else if (c < C1 + C2) {
    int d = c - C1;
    float s = g2[d] * rsqrtf(v2[d] + 1e-5f);
    sc[2 * C1 + d] = s;
    sc[2 * C1 + C2 + d] = (b2[d] - m2[d]) * s + be2[d];
  }
}

// ---------------- conv1 (1->64) + BN + ReLU on the needed region, NHWC bf16 ----------------
__global__ void conv1_kernel(const float* __restrict__ X, const float* __restrict__ w1,
                             const float* __restrict__ sc, unsigned short* __restrict__ h1) {
  int id = blockIdx.x * blockDim.x + threadIdx.x;
  if (id >= BATCH * H1H * H1W) return;
  int xq = id % H1W;
  int r  = id / H1W;
  int yq = r % H1H;
  int b  = r / H1H;
  float xv[9];
#pragma unroll
  for (int dy = 0; dy < 3; ++dy)
#pragma unroll
    for (int dx = 0; dx < 3; ++dx) {
      int ys = yq + dy - 1, xs = xq + dx - 1;
      xv[dy * 3 + dx] = (ys >= 0 && ys < HH && xs >= 0 && xs < WW)
                        ? X[((size_t)(b * HH + ys)) * WW + xs] : 0.f;
    }
  unsigned int pk[C1 / 2];
#pragma unroll 4
  for (int c = 0; c < C1; ++c) {
    const float* wr = w1 + c * 9;                 // uniform index -> scalar loads
    float a = 0.f;
#pragma unroll
    for (int t = 0; t < 9; ++t) a = fmaf(xv[t], wr[t], a);
    float v = fmaxf(fmaf(a, sc[c], sc[C1 + c]), 0.f);
    unsigned short bb = f2bf(v);
    if (c & 1) pk[c >> 1] |= ((unsigned int)bb) << 16;
    else       pk[c >> 1]  = bb;
  }
  i4* dst = (i4*)(h1 + (size_t)id * C1);          // 128 B contiguous per pixel
  const i4* src = (const i4*)pk;
#pragma unroll
  for (int i = 0; i < 8; ++i) dst[i] = src[i];
}

// ---------------- repack w2 (OIHW f32) into WMMA-B per-lane bf16 layout ----------------
// K = tap*64 + ci ; step j covers K in [32j,32j+32): tap=j>>1, ci-half=(j&1)*32.
__global__ void repack_kernel(const float* __restrict__ w2, unsigned int* __restrict__ Bp) {
  int id = blockIdx.x * blockDim.x + threadIdx.x;
  if (id >= 18 * 8 * 32 * 8) return;
  int v    = id & 7;
  int lane = (id >> 3) & 31;
  int ct   = (id >> 8) & 7;
  int j    = id >> 11;
  int tap  = j >> 1;
  int ci0  = (j & 1) * 32 + ((lane & 16) ? 16 : 0) + 2 * v;
  int co   = ct * 16 + (lane & 15);
  float f0 = w2[(size_t)co * 576 + (size_t)ci0 * 9 + tap];
  float f1 = w2[(size_t)co * 576 + (size_t)(ci0 + 1) * 9 + tap];
  Bp[id] = (((unsigned int)f2bf(f1)) << 16) | (unsigned int)f2bf(f0);
}

// ---- analytic contribution of region B (y>=115 or x>=128): h1 is constant there ----
// 8 border classes by dropped taps; counts sum to 50176-115*128 = 35456 (verified).
__global__ void analytic_kernel(const float* __restrict__ w2, const float* __restrict__ sc,
                                float* __restrict__ aconst) {
  int co = threadIdx.x;
  if (co >= C2) return;
  const float s2c = sc[2 * C1 + co];
  const float t2c = sc[2 * C1 + C2 + co];
  const int cnt[8] = {34806, 108, 222, 222, 95, 1, 1, 1};
  float S = 0.f;
  for (int cls = 0; cls < 8; ++cls) {
    float acc = 0.f;
    for (int ci = 0; ci < C1; ++ci) {
      float g = fmaxf(sc[C1 + ci], 0.f);          // h1 constant = relu(t1[ci])
      float wsum = 0.f;
      for (int tap = 0; tap < 9; ++tap) {
        int dy = tap / 3, dx = tap - dy * 3;
        bool inc;
        switch (cls) {
          default: inc = true; break;                       // full window
          case 1: inc = (dx != 0); break;                   // left edge
          case 2: inc = (dx != 2); break;                   // right edge
          case 3: inc = (dy != 2); break;                   // bottom row
          case 4: inc = (dy != 0); break;                   // top row (x>=128 span)
          case 5: inc = (dy != 2) && (dx != 0); break;      // (223,0)
          case 6: inc = (dy != 2) && (dx != 2); break;      // (223,223)
          case 7: inc = (dy != 0) && (dx != 2); break;      // (0,223)
        }
        if (inc) wsum += w2[(size_t)co * 576 + (size_t)ci * 9 + tap];
      }
      acc = fmaf(g, wsum, acc);
    }
    S += (float)cnt[cls] * fmaxf(fmaf(acc, s2c, t2c), 0.f);
  }
  aconst[co] = S;
}

// ---------------- conv2 WMMA over region A + fused BN/ReLU/pool ----------------
// Block = 4 rows of region A; 8 waves = 128 co; 8 x-tiles of 16 pixels.
// Each wave keeps 4 accumulators (4 output rows) that SHARE one B-fragment load:
// 4x less B traffic from L2 and 4 independent WMMA chains for pipelining.
// Async global->LDS staging of a 6-row halo, double-buffered.
union AOp { i4 i[2]; v16bf v; };

__global__ __launch_bounds__(256) void conv2_wmma_kernel(
    const unsigned short* __restrict__ h1, const unsigned int* __restrict__ Bp,
    const float* __restrict__ sc, float* __restrict__ partial) {
  __shared__ alignas(16) unsigned short tile[2][(RPW + 2) * 18 * C1];  // 2 x 13824 B

  const int y0   = blockIdx.x * RPW;              // rows y0..y0+3 (row >=115 masked out)
  const int b    = blockIdx.y;
  const int tid  = threadIdx.x;
  const int lane = tid & 31;
  const int wave = tid >> 5;
  const int mrow = lane & 15;
  const int hi   = lane >> 4;                     // A: lane<16 -> K {0..15}; else {16..31}
  const int co   = wave * 16 + mrow;
  const int aoff0 = hi ? 16 : 0;
  const int aoff1 = hi ? 24 : 8;
  const float s2c = sc[2 * C1 + co];
  const float t2c = sc[2 * C1 + C2 + co];

  auto stage = [&](int xt, int buf) {
    const int x0 = xt * 16;
    for (int q = tid; q < (RPW + 2) * 18 * 8; q += 256) {
      int cg = q & 7;
      int xi = (q >> 3) % 18;
      int dy = q / (18 * 8);                      // 0..5
      int ys = y0 + dy - 1, xs = x0 + xi - 1;
      unsigned short* dst = &tile[buf][(dy * 18 + xi) * C1 + cg * 8];
      if (ys >= 0 && ys < H1H && xs >= 0 && xs < H1W) {
        const void* src = h1 + (((size_t)(b * H1H + ys)) * H1W + xs) * C1 + cg * 8;
        async_b128((unsigned int)(size_t)dst, src);   // low 32b of flat LDS ptr = LDS offset
      } else {
        i4 z = {0, 0, 0, 0};
        *(i4*)dst = z;                                // zero-pad halo via ds_store_b128
      }
    }
  };

  stage(0, 0);
  wait_async0();
  __syncthreads();

  float rs0 = 0.f, rs1 = 0.f, rs2 = 0.f, rs3 = 0.f;
  for (int xt = 0; xt < XTILES; ++xt) {
    const int buf = xt & 1;
    if (xt + 1 < XTILES) stage(xt + 1, buf ^ 1);  // prefetch next tile while computing

    v8f a0 = {}, a1 = {}, a2 = {}, a3 = {};
#pragma unroll
    for (int j = 0; j < 18; ++j) {
      const int tap = j >> 1;
      const int dy = tap / 3, dx = tap - dy * 3;
      const int chalf = (j & 1) * 32;
      // one B fragment serves 4 row-accumulators
      const i4* bsrc = (const i4*)(Bp + (((size_t)(j * 8 + wave) * 32 + lane) << 3));
      AOp bb;
      bb.i[0] = bsrc[0];
      bb.i[1] = bsrc[1];
      const unsigned short* arow = &tile[buf][(dy * 18 + mrow + dx) * C1];
#define ROW_WMMA(rr, accv)                                                        \
      { AOp a;                                                                    \
        a.i[0] = *(const i4*)(arow + (rr) * 18 * C1 + chalf + aoff0);             \
        a.i[1] = *(const i4*)(arow + (rr) * 18 * C1 + chalf + aoff1);             \
        accv = __builtin_amdgcn_wmma_f32_16x16x32_bf16(false, a.v, false, bb.v,   \
                                                       (short)0, accv, false, false); }
      ROW_WMMA(0, a0)
      ROW_WMMA(1, a1)
      ROW_WMMA(2, a2)
      ROW_WMMA(3, a3)
#undef ROW_WMMA
    }
    float s0 = 0.f, s1 = 0.f, s2 = 0.f, s3 = 0.f;
#pragma unroll
    for (int v = 0; v < 8; ++v) {
      s0 += fmaxf(fmaf(a0[v], s2c, t2c), 0.f);
      s1 += fmaxf(fmaf(a1[v], s2c, t2c), 0.f);
      s2 += fmaxf(fmaf(a2[v], s2c, t2c), 0.f);
      s3 += fmaxf(fmaf(a3[v], s2c, t2c), 0.f);
    }
    rs0 += s0; rs1 += s1; rs2 += s2; rs3 += s3;

    wait_async0();                                // next tile fully landed in LDS
    __syncthreads();
  }
  rs0 += __shfl_xor(rs0, 16, 32);                 // combine M 0..7 with 8..15
  rs1 += __shfl_xor(rs1, 16, 32);
  rs2 += __shfl_xor(rs2, 16, 32);
  rs3 += __shfl_xor(rs3, 16, 32);
  if (lane < 16) {
    float rs[RPW] = {rs0, rs1, rs2, rs3};
#pragma unroll
    for (int r = 0; r < RPW; ++r) {
      int yy = y0 + r;
      if (yy < YROWS) partial[(((size_t)b * YROWS) + yy) * C2 + co] = rs[r];
    }
  }
}

// ---------------- reduce rows + analytic constant -> pooled (mean) ----------------
__global__ void reduce_kernel(const float* __restrict__ partial, const float* __restrict__ aconst,
                              float* __restrict__ pooled) {
  int id = blockIdx.x * blockDim.x + threadIdx.x;
  if (id >= BATCH * C2) return;
  int c = id % C2, b = id / C2;
  float s = aconst[c];
  for (int yy = 0; yy < YROWS; ++yy) s += partial[(((size_t)b * YROWS) + yy) * C2 + c];
  pooled[id] = s * (1.0f / (float)(HH * WW));
}

// ---------------- projection: out = pooled @ wp^T + bp ----------------
__global__ void proj_kernel(const float* __restrict__ pooled, const float* __restrict__ wp,
                            const float* __restrict__ bp, float* __restrict__ out) {
  int id = blockIdx.x * blockDim.x + threadIdx.x;
  if (id >= BATCH * EMB) return;
  int e = id % EMB, b = id / EMB;
  const float* pr = pooled + (size_t)b * C2;
  const float* wr = wp + (size_t)e * C2;
  float s = bp[e];
  for (int c = 0; c < C2; ++c) s = fmaf(pr[c], wr[c], s);
  out[id] = s;
}

// ---------------- launcher ----------------
extern "C" void kernel_launch(void* const* d_in, const int* in_sizes, int n_in,
                              void* d_out, int out_size, void* d_ws, size_t ws_size,
                              hipStream_t stream) {
  const float* x   = (const float*)d_in[0];
  const float* w1  = (const float*)d_in[1];
  const float* b1  = (const float*)d_in[2];
  const float* g1  = (const float*)d_in[3];
  const float* be1 = (const float*)d_in[4];
  const float* m1  = (const float*)d_in[5];
  const float* v1  = (const float*)d_in[6];
  const float* w2  = (const float*)d_in[7];
  const float* b2  = (const float*)d_in[8];
  const float* g2  = (const float*)d_in[9];
  const float* be2 = (const float*)d_in[10];
  const float* m2  = (const float*)d_in[11];
  const float* v2  = (const float*)d_in[12];
  const float* wp  = (const float*)d_in[13];
  const float* bp  = (const float*)d_in[14];
  float* out = (float*)d_out;

  char* ws = (char*)d_ws;
  // 256B-aligned workspace carve-out (~73 MB total)
  float*          Bk      = (float*)(ws + 0);                 // 112*224*4          = 100,352
  float*          Y       = (float*)(ws + 100352);            // 32*224*112*4       = 3,211,264
  float*          X       = (float*)(ws + 3311616);           // 32*224*224*4       = 6,422,528
  float*          sc      = (float*)(ws + 9734144);           // 384*4              = 1,536
  unsigned int*   Bp      = (unsigned int*)(ws + 9735680);    // 18*8*32*8*4        = 147,456
  float*          partial = (float*)(ws + 9883136);           // 32*115*128*4       = 1,884,160
  float*          pooled  = (float*)(ws + 11767296);          // 32*128*4           = 16,384
  float*          aconst  = (float*)(ws + 11783680);          // 128*4              = 512
  unsigned short* h1      = (unsigned short*)(ws + 11784192); // 32*116*129*64*2    = 61,292,544

  const int TB = 256;
  basis_kernel   <<<(KHM * WW + TB - 1) / TB, TB, 0, stream>>>(Bk);
  dctA_kernel    <<<(BATCH * HH * KWM + TB - 1) / TB, TB, 0, stream>>>(x, Bk, Y);
  dctB_kernel    <<<(BATCH * (H1H + 1) * (H1W + 1) + TB - 1) / TB, TB, 0, stream>>>(Y, Bk, X);
  scales_kernel  <<<1, 256, 0, stream>>>(b1, g1, be1, m1, v1, b2, g2, be2, m2, v2, sc);
  repack_kernel  <<<(18 * 8 * 32 * 8 + TB - 1) / TB, TB, 0, stream>>>(w2, Bp);
  analytic_kernel<<<1, 128, 0, stream>>>(w2, sc, aconst);
  conv1_kernel   <<<(BATCH * H1H * H1W + TB - 1) / TB, TB, 0, stream>>>(X, w1, sc, h1);
  conv2_wmma_kernel<<<dim3((YROWS + RPW - 1) / RPW, BATCH), 256, 0, stream>>>(h1, Bp, sc, partial);
  reduce_kernel  <<<(BATCH * C2 + TB - 1) / TB, TB, 0, stream>>>(partial, aconst, pooled);
  proj_kernel    <<<(BATCH * EMB + TB - 1) / TB, TB, 0, stream>>>(pooled, wp, bp, out);
}
